// FactoredDynamicsModel_46858093199484
// MI455X (gfx1250) — matrix operand order repaired and checked by reading, not compiled, not run on hardware
//
#include <hip/hip_runtime.h>
#include <hip/hip_bf16.h>

// ---------------------------------------------------------------------------
// FactoredDynamicsModel on gfx1250 (MI455X), f16 WMMA path.
//   unary:  h = relu(x@uW1+b) -> relu(h@uW2+b) -> h@uW3+b           (65536 rows)
//   binary: h1[b,i,j] = relu(U[b,i]+V[b,j]+D[b])   (rank-split of W1)
//           h2 = relu(h1@bW2+bb2)  summed over j (diag masked)
//           b_sum = (sum_j h2)@bW3 + 15*bb3        (layer-3 pulled out of sum)
// Dominant GEMM: 1M x 128 x 128 on v_wmma_f32_16x16x32_f16.
// LDS staging uses GLOBAL_LOAD_ASYNC_TO_LDS_B128 (ASYNCcnt) when available.
// ---------------------------------------------------------------------------

typedef __attribute__((ext_vector_type(16))) _Float16 v16h;
typedef __attribute__((ext_vector_type(8)))  _Float16 v8h;
typedef __attribute__((ext_vector_type(8)))  float    v8f;
typedef int v4i __attribute__((vector_size(16)));

#define HID 128
#define STP 136   // padded LDS stride (halfs): 272B rows -> conflict-free b128 frag loads
                  // 136 halfs = 17 x 16B chunks exactly

#if defined(__gfx1250__) && __has_builtin(__builtin_amdgcn_global_load_async_to_lds_b128)
#define USE_ASYNC 1
#else
#define USE_ASYNC 0
#endif

#if USE_ASYNC
typedef __attribute__((address_space(3))) v4i as3_v4i;
#endif

// async (or fallback) 16-byte global -> LDS copy
__device__ __forceinline__ void lds_copy16(const _Float16* gsrc, _Float16* ldst) {
#if USE_ASYNC
  __builtin_amdgcn_global_load_async_to_lds_b128((v4i*)gsrc, (as3_v4i*)ldst, 0, 0);
#else
  *(uint4*)ldst = *(const uint4*)gsrc;
#endif
}
__device__ __forceinline__ void lds_copy_wait() {
#if USE_ASYNC
#if __has_builtin(__builtin_amdgcn_s_wait_asynccnt)
  __builtin_amdgcn_s_wait_asynccnt(0);
#else
  asm volatile("s_wait_asynccnt 0x0" ::: "memory");
#endif
#endif
}

union V16H { v16h v; struct { v8h lo, hi; } p; };

// A operand (16x32 f16): lane = row M; v0-3 hold K = half*8 + 0..7, v4-7 hold +16.
__device__ __forceinline__ v16h wmma_load_A(const _Float16* tile, int stride, int k0, int lane) {
  int row = lane & 15, half = lane >> 4;
  const _Float16* p = tile + row * stride + k0 + half * 8;
  V16H r; r.p.lo = *(const v8h*)p; r.p.hi = *(const v8h*)(p + 16);
  return r.v;
}
// B operand (32x16 f16) from K-contiguous (transposed) weights: lane = col N,
// v0-3 hold K = half*8 + 0..7, v4-7 hold +16.
__device__ __forceinline__ v16h wmma_load_B(const _Float16* wt, int stride, int n0, int k0, int lane) {
  int col = n0 + (lane & 15), half = lane >> 4;
  const _Float16* p = wt + col * stride + k0 + half * 8;
  V16H r; r.p.lo = *(const v8h*)p; r.p.hi = *(const v8h*)(p + 16);
  return r.v;
}
__device__ __forceinline__ v8f wmma16(v16h a, v16h b, v8f c) {
  return __builtin_amdgcn_wmma_f32_16x16x32_f16(false, a, false, b, (short)0, c, false, false);
}

// Build A-fragment of h1 = relu(U[i] + V[j] + D) directly in registers (packed f16 math).
__device__ __forceinline__ v16h make_h1_frag(const _Float16* sU, const _Float16* sV,
                                             const _Float16* sDh, int i, int k0, int lane) {
  int j = lane & 15, half = lane >> 4;
  int kb0 = k0 + half * 8;
  int kb1 = kb0 + 16;
  v8h z = {};
  v8h x0 = *(const v8h*)(sU + i * STP + kb0) + *(const v8h*)(sV + j * STP + kb0)
         + *(const v8h*)(sDh + kb0);
  v8h x1 = *(const v8h*)(sU + i * STP + kb1) + *(const v8h*)(sV + j * STP + kb1)
         + *(const v8h*)(sDh + kb1);
  x0 = __builtin_elementwise_max(x0, z);
  x1 = __builtin_elementwise_max(x1, z);
  V16H r; r.p.lo = x0; r.p.hi = x1;
  return r.v;
}

// ---------------------------------------------------------------- prep kernels
// Writes transposed f16 weights ALREADY padded to the STP stride, so LDS staging
// is a pure linear 16B-chunk stream for the async-to-LDS engine.
extern "C" __global__ void fdm_prep_weights(const float* bW2, const float* bW3,
                                            const float* uW2, const float* uW3,
                                            _Float16* W2Tb, _Float16* W3Tb,
                                            _Float16* W2Tu, _Float16* W3Tu) {
  int t = blockIdx.x * blockDim.x + threadIdx.x;          // 17408 threads
  if (t < 128 * STP) {
    int n = t / STP, k = t % STP;
    _Float16 vb = (_Float16)0.f, vu = (_Float16)0.f;
    if (k < 128) { vb = (_Float16)bW2[k * 128 + n]; vu = (_Float16)uW2[k * 128 + n]; }
    W2Tb[t] = vb; W2Tu[t] = vu;                           // [N][STP] K-contiguous
  }
  if (t < 16 * STP) {
    int n = t / STP, k = t % STP;
    float vb = 0.f, vu = 0.f;
    if (k < 128 && n < 6) { vb = bW3[k * 6 + n]; vu = uW3[k * 6 + n]; }  // pad N 6->16
    W3Tb[t] = (_Float16)vb; W3Tu[t] = (_Float16)vu;
  }
}

extern "C" __global__ void fdm_prep_per_b(const float* action, const float* bW1, const float* uW1,
                                          const float* bb1, const float* ub1,
                                          float* Db, float* Qu) {
  int t = blockIdx.x * blockDim.x + threadIdx.x;          // 4096*128
  int b = t >> 7, h = t & 127;
  float sa = 0.f, su = 0.f;
  #pragma unroll 8
  for (int a = 0; a < 32; ++a) {
    float av = action[b * 32 + a];
    sa += av * bW1[(12 + a) * 128 + h];
    su += av * uW1[(6 + a) * 128 + h];
  }
  Db[t] = sa + bb1[h];
  Qu[t] = su + ub1[h];
}

extern "C" __global__ void fdm_prep_per_r(const float* cfg, const float* bW1, const float* uW1,
                                          const float* Qu, _Float16* U, _Float16* V,
                                          _Float16* h1u) {
  int t = blockIdx.x * blockDim.x + threadIdx.x;          // 65536*128
  int r = t >> 7, h = t & 127;
  float su = 0.f, sv = 0.f, sq = 0.f;
  #pragma unroll
  for (int d = 0; d < 6; ++d) {
    float o = cfg[r * 6 + d];
    su += o * bW1[d * 128 + h];
    sv += o * bW1[(6 + d) * 128 + h];
    sq += o * uW1[d * 128 + h];
  }
  U[t] = (_Float16)su;
  V[t] = (_Float16)sv;
  float q = sq + Qu[(r >> 4) * 128 + h];
  h1u[t] = (_Float16)(q > 0.f ? q : 0.f);
}

// ---------------------------------------------------------------- unary MLP
#define U_IT 4   // row-tiles per block (amortize weight staging)
extern "C" __global__ void __launch_bounds__(128)
fdm_unary_kernel(const _Float16* __restrict__ h1u, const _Float16* __restrict__ W2T,
                 const _Float16* __restrict__ W3T, const float* __restrict__ ub2,
                 const float* __restrict__ ub3, float* __restrict__ u_out) {
  __shared__ alignas(16) _Float16 sW2[128 * STP];
  __shared__ alignas(16) _Float16 sW3[16 * STP];
  __shared__ alignas(16) _Float16 sH2[4 * 16 * STP];
  __shared__ float sB2[HID];

  const int tid = threadIdx.x;
  const int wave = tid >> 5, lane = tid & 31;

  // stage padded weights: linear async 16B-chunk stream into LDS
  for (int c = tid; c < 128 * 17; c += 128) lds_copy16(W2T + c * 8, sW2 + c * 8);
  for (int c = tid; c < 16 * 17;  c += 128) lds_copy16(W3T + c * 8, sW3 + c * 8);
  if (tid < HID) sB2[tid] = ub2[tid];
  lds_copy_wait();
  __syncthreads();

  const int half = lane >> 4, col = lane & 15;
  _Float16* h2 = sH2 + wave * 16 * STP;

  for (int it = 0; it < U_IT; ++it) {
    const int r0 = (blockIdx.x * U_IT + it) * 64 + wave * 16;

    v16h af[4];
    #pragma unroll
    for (int k = 0; k < 4; ++k)
      af[k] = wmma_load_A(h1u + (size_t)r0 * HID, HID, k * 32, lane);  // global b128 frags

    #pragma unroll
    for (int n = 0; n < 8; ++n) {                  // layer 2: 8 N-tiles x 4 K-steps
      v8f acc = {};
      #pragma unroll
      for (int k = 0; k < 4; ++k)
        acc = wmma16(af[k], wmma_load_B(sW2, STP, n * 16, k * 32, lane), acc);
      float bias = sB2[n * 16 + col];
      #pragma unroll
      for (int v = 0; v < 8; ++v) {
        float x = acc[v] + bias; x = x > 0.f ? x : 0.f;
        h2[(v + half * 8) * STP + n * 16 + col] = (_Float16)x;
      }
    }
    __syncthreads();

    v8f a3 = {};                                   // layer 3 (N padded to 16)
    #pragma unroll
    for (int k = 0; k < 4; ++k)
      a3 = wmma16(wmma_load_A(h2, STP, k * 32, lane), wmma_load_B(sW3, STP, 0, k * 32, lane), a3);
    if (col < 6) {
      float b3 = ub3[col];
      #pragma unroll
      for (int v = 0; v < 8; ++v)
        u_out[(size_t)(r0 + v + half * 8) * 8 + col] = a3[v] + b3;
    }
    __syncthreads();                               // protect sH2 before next tile
  }
}

// ---------------------------------------------------------------- binary MLP (dominant)
#define BPB 4    // batch elements per block (weights staged once)
extern "C" __global__ void __launch_bounds__(256)
fdm_binary_kernel(const _Float16* __restrict__ Ug, const _Float16* __restrict__ Vg,
                  const float* __restrict__ Db, const _Float16* __restrict__ W2T,
                  const _Float16* __restrict__ W3T, const float* __restrict__ bb2,
                  const float* __restrict__ bb3, const float* __restrict__ cfg,
                  const float* __restrict__ u_out, float* __restrict__ out) {
  __shared__ alignas(16) _Float16 sW2T[128 * STP];
  __shared__ alignas(16) _Float16 sW3T[16 * STP];
  __shared__ alignas(16) _Float16 sU[16 * STP];
  __shared__ alignas(16) _Float16 sV[16 * STP];
  __shared__ alignas(16) _Float16 sDh[HID];
  __shared__ alignas(16) _Float16 sS[16 * STP];
  __shared__ float sB2[HID];

  const int b0 = blockIdx.x * BPB;
  const int tid = threadIdx.x;
  const int wave = tid >> 5, lane = tid & 31;
  const int half = lane >> 4;

  // stage padded weights once: linear async 16B-chunk stream
  for (int c = tid; c < 128 * 17; c += 256) lds_copy16(W2T + c * 8, sW2T + c * 8);
  for (int c = tid; c < 16 * 17;  c += 256) lds_copy16(W3T + c * 8, sW3T + c * 8);
  if (tid < HID) sB2[tid] = bb2[tid];

  for (int it = 0; it < BPB; ++it) {
    const int b = b0 + it;
    // stage U,V rows (16 x 128 halfs each = 256 chunks each) with stride padding
    {
      int n = tid >> 4, q = tid & 15;             // 256 threads = 256 chunks
      lds_copy16(Ug + (size_t)b * 16 * HID + n * HID + q * 8, sU + n * STP + q * 8);
      lds_copy16(Vg + (size_t)b * 16 * HID + n * HID + q * 8, sV + n * STP + q * 8);
    }
    if (tid < HID) sDh[tid] = (_Float16)Db[(size_t)b * HID + tid];
    lds_copy_wait();
    __syncthreads();

    // wave owns groups i = wave and wave+8 (rows of a group = j = 0..15)
    const int iA = wave, iB = wave + 8;
    v16h afA[4], afB[4];
    #pragma unroll
    for (int k = 0; k < 4; ++k) {
      afA[k] = make_h1_frag(sU, sV, sDh, iA, k * 32, lane);
      afB[k] = make_h1_frag(sU, sV, sDh, iB, k * 32, lane);
    }

    #pragma unroll
    for (int n = 0; n < 8; ++n) {                  // layer 2: B-frag shared by 2 M-tiles
      v8f aA = {}, aB = {};
      #pragma unroll
      for (int k = 0; k < 4; ++k) {
        v16h bf = wmma_load_B(sW2T, STP, n * 16, k * 32, lane);
        aA = wmma16(afA[k], bf, aA);
        aB = wmma16(afB[k], bf, aB);
      }
      float bias = sB2[n * 16 + (lane & 15)];
      float sA = 0.f, sB = 0.f;                    // bias + relu + diag mask + sum over j
      #pragma unroll
      for (int v = 0; v < 8; ++v) {
        int row = v + half * 8;
        float xA = aA[v] + bias; xA = xA > 0.f ? xA : 0.f; if (row == iA) xA = 0.f; sA += xA;
        float xB = aB[v] + bias; xB = xB > 0.f ? xB : 0.f; if (row == iB) xB = 0.f; sB += xB;
      }
      sA += __shfl_xor(sA, 16);
      sB += __shfl_xor(sB, 16);
      if (lane < 16) {
        sS[iA * STP + n * 16 + lane] = (_Float16)sA;
        sS[iB * STP + n * 16 + lane] = (_Float16)sB;
      }
    }
    __syncthreads();

    if (wave == 0) {                               // layer 3 on summed h2 + final fuse
      v8f acc = {};
      #pragma unroll
      for (int k = 0; k < 4; ++k)
        acc = wmma16(wmma_load_A(sS, STP, k * 32, lane),
                     wmma_load_B(sW3T, STP, 0, k * 32, lane), acc);
      int d = lane & 15;
      if (d < 6) {
        float b3 = 15.0f * bb3[d];                 // 15 off-diagonal bias terms
        #pragma unroll
        for (int v = 0; v < 8; ++v) {
          int i = v + half * 8;
          size_t r = (size_t)b * 16 + i;
          out[(size_t)b * 96 + i * 6 + d] =
              acc[v] + b3 + u_out[r * 8 + d] + cfg[(size_t)b * 96 + i * 6 + d];
        }
      }
    }
    __syncthreads();                               // protect sU/sV/sS before next batch
  }
}

// ---------------------------------------------------------------- launch
extern "C" void kernel_launch(void* const* d_in, const int* in_sizes, int n_in,
                              void* d_out, int out_size, void* d_ws, size_t ws_size,
                              hipStream_t stream) {
  (void)in_sizes; (void)n_in; (void)out_size; (void)ws_size;
  const float* cfg    = (const float*)d_in[0];
  const float* action = (const float*)d_in[1];
  const float* uW1 = (const float*)d_in[2];
  const float* ub1 = (const float*)d_in[3];
  const float* uW2 = (const float*)d_in[4];
  const float* ub2 = (const float*)d_in[5];
  const float* uW3 = (const float*)d_in[6];
  const float* ub3 = (const float*)d_in[7];
  const float* bW1 = (const float*)d_in[8];
  const float* bb1 = (const float*)d_in[9];
  const float* bW2 = (const float*)d_in[10];
  const float* bb2 = (const float*)d_in[11];
  const float* bW3 = (const float*)d_in[12];
  const float* bb3 = (const float*)d_in[13];

  // workspace layout (~56.7 MB total), all offsets 128B-aligned
  char* ws = (char*)d_ws;
  _Float16* W2Tb = (_Float16*)(ws);                 // 128*136 f16 = 34816 B (padded)
  _Float16* W3Tb = (_Float16*)(ws + 34816);         // 16*136 f16  = 4352 B
  _Float16* W2Tu = (_Float16*)(ws + 39168);         // 34816 B
  _Float16* W3Tu = (_Float16*)(ws + 73984);         // 4352 B
  float*    Db   = (float*)(ws + 78336);            // 4096*128 f32 = 2 MB
  float*    Qu   = (float*)(ws + 2175488);          // 2 MB
  _Float16* U    = (_Float16*)(ws + 4272640);       // 65536*128 f16 = 16 MB
  _Float16* V    = (_Float16*)(ws + 21049856);      // 16 MB
  _Float16* h1u  = (_Float16*)(ws + 37827072);      // 16 MB
  float*    u_o  = (float*)(ws + 54604288);         // 65536*8 f32 = 2 MB

  fdm_prep_weights<<<68, 256, 0, stream>>>(bW2, bW3, uW2, uW3, W2Tb, W3Tb, W2Tu, W3Tu);
  fdm_prep_per_b<<<2048, 256, 0, stream>>>(action, bW1, uW1, bb1, ub1, Db, Qu);
  fdm_prep_per_r<<<32768, 256, 0, stream>>>(cfg, bW1, uW1, Qu, U, V, h1u);
  fdm_unary_kernel<<<256, 128, 0, stream>>>(h1u, W2Tu, W3Tu, ub2, ub3, u_o);
  fdm_binary_kernel<<<1024, 256, 0, stream>>>(U, V, Db, W2Tb, W3Tb, bb2, bb3, cfg, u_o,
                                              (float*)d_out);
}
